// GCN_S2S_Extractor_36799279792904
// MI455X (gfx1250) — matrix operand order repaired and checked by reading, not compiled, not run on hardware
//
#include <hip/hip_runtime.h>
#include <hip/hip_bf16.h>
#include <math.h>

// ---------------------------------------------------------------------------
// GCN (3 layers) + Set2Set + DNN head, two branches, MI455X (gfx1250).
// Dense transforms use v_wmma_f32_16x16x32_bf16 (f32 accumulate).
// Edge aggregation = coalesced gather + global_atomic_add_f32 scatter.
// Train-mode BN absorbs the per-channel GCN/dnn1 biases (constant per row
// shifts cancel in (x - mean)), so those biases are mathematically dropped.
// ---------------------------------------------------------------------------

typedef __attribute__((ext_vector_type(16))) __bf16 v16bf;
typedef __attribute__((ext_vector_type(8)))  __bf16 v8bf;
typedef __attribute__((ext_vector_type(8)))  float  v8f;

#define CDIV(a, b) (((a) + (b) - 1) / (b))

__device__ __forceinline__ float sigmoidf_(float x) { return 1.0f / (1.0f + expf(-x)); }
__device__ __forceinline__ unsigned fOrd(float f) {
  unsigned u = __float_as_uint(f);
  return (u & 0x80000000u) ? ~u : (u | 0x80000000u);
}
__device__ __forceinline__ float fUnord(unsigned u) {
  return __uint_as_float((u & 0x80000000u) ? (u & 0x7fffffffu) : ~u);
}

// ------------------------------- utility -----------------------------------
__global__ void fill_f_k(float* p, float v, long long n) {
  long long i = (long long)blockIdx.x * blockDim.x + threadIdx.x;
  if (i < n) p[i] = v;
}
__global__ void fill_u_k(unsigned* p, unsigned v, long long n) {
  long long i = (long long)blockIdx.x * blockDim.x + threadIdx.x;
  if (i < n) p[i] = v;
}

// ------------------------------ gcn_norm -----------------------------------
__global__ void deg_edges_k(const long long* __restrict__ ei,
                            const float* __restrict__ w,
                            float* __restrict__ deg, int E) {
  int e = blockIdx.x * blockDim.x + threadIdx.x;
  if (e >= E) return;
  atomicAdd(&deg[(int)ei[(long long)E + e]], w[e]);
}
__global__ void dinv_k(const float* __restrict__ deg, float* __restrict__ dinv, int n) {
  int i = blockIdx.x * blockDim.x + threadIdx.x;
  if (i >= n) return;
  float d = deg[i];
  dinv[i] = d > 0.f ? rsqrtf(fmaxf(d, 1e-12f)) : 0.f;
}
__global__ void norm_k(const long long* __restrict__ ei, const float* __restrict__ w,
                       const float* __restrict__ dinv, float* __restrict__ nrm,
                       int E, int N) {
  int e = blockIdx.x * blockDim.x + threadIdx.x;
  if (e >= E + N) return;
  if (e < E)
    nrm[e] = dinv[(int)ei[e]] * w[e] * dinv[(int)ei[(long long)E + e]];
  else {
    int n = e - E;
    nrm[e] = dinv[n] * dinv[n];  // self-loop, weight 1
  }
}

// ------------------------------ batchnorm ----------------------------------
// blockDim.x == C; each block accumulates a row chunk.
__global__ void bn_stats_k(const float* __restrict__ X, int N, int C,
                           float* __restrict__ sum, float* __restrict__ sq,
                           int rowsPerBlock) {
  int c = threadIdx.x;
  int r0 = blockIdx.x * rowsPerBlock;
  int r1 = min(r0 + rowsPerBlock, N);
  float s = 0.f, q = 0.f;
  for (int r = r0; r < r1; ++r) {
    float v = X[(size_t)r * C + c];
    s += v;
    q += v * v;
  }
  atomicAdd(&sum[c], s);
  atomicAdd(&sq[c], q);
}
__global__ void bn_finalize_k(const float* __restrict__ sum, const float* __restrict__ sq,
                              const float* __restrict__ gamma, const float* __restrict__ beta,
                              float* __restrict__ scale, float* __restrict__ shift,
                              int N, int C) {
  int c = blockIdx.x * blockDim.x + threadIdx.x;
  if (c >= C) return;
  float mu = sum[c] / N;
  float var = sq[c] / N - mu * mu;
  float sc = gamma[c] * rsqrtf(var + 1e-5f);
  scale[c] = sc;
  shift[c] = beta[c] - mu * sc;
}
// y = x*scale + shift, optional ELU, write bf16 and/or f32.
__global__ void bn_apply_k(const float* __restrict__ X, const float* __restrict__ scale,
                           const float* __restrict__ shift, __bf16* outBf, float* outF,
                           long long total, int C, int doElu) {
  long long i = (long long)blockIdx.x * blockDim.x + threadIdx.x;
  if (i >= total) return;
  int c = (int)(i % C);
  float v = X[i] * scale[c] + shift[c];
  if (doElu) v = v > 0.f ? v : expm1f(v);
  if (outBf) outBf[i] = (__bf16)v;
  if (outF) outF[i] = v;
}

// --------------------------- WMMA dense GEMM -------------------------------
// Wt[c*K + k] = bf16(W[k*Co + c])  (transposed copy for contiguous K loads)
__global__ void conv_wt_k(const float* __restrict__ W, __bf16* __restrict__ Wt,
                          int K, int Co) {
  int i = blockIdx.x * blockDim.x + threadIdx.x;
  if (i >= K * Co) return;
  int k = i / Co, c = i % Co;
  Wt[(size_t)c * K + k] = (__bf16)W[i];
}

// H(N x Co) = Xbf(N x K) @ Wt^T, bf16 inputs, f32 accumulate via WMMA.
// One wave per 16x16 output tile, 8 waves / block. K,Co multiples of 32/16.
__global__ __launch_bounds__(256) void wmma_mm_k(const __bf16* __restrict__ X,
                                                 const __bf16* __restrict__ Wt,
                                                 float* __restrict__ H,
                                                 int N, int K, int Co) {
  int wave = threadIdx.x >> 5;
  int lane = threadIdx.x & 31;
  int tilesN = Co >> 4;
  long long tile = (long long)blockIdx.x * 8 + wave;
  long long totalTiles = (long long)(N >> 4) * tilesN;
  if (tile >= totalTiles) return;  // wave-uniform: EXEC stays all-ones
  int m0 = (int)(tile / tilesN) << 4;
  int n0 = (int)(tile % tilesN) << 4;

  int row = m0 + (lane & 15);          // A: row per lane
  int colB = n0 + (lane & 15);         // B: column per lane
  int kh = (lane >> 4) << 3;           // A half: K offset 0 or 8
  int kb = (lane >> 4) << 3;           // B half: K offset 0 or 8

  v8f acc = {};
  for (int k0 = 0; k0 < K; k0 += 32) {
    const __bf16* pa = X + (size_t)row * K + k0 + kh;
    v8bf alo = *(const v8bf*)pa;         // K = k0+kh .. +7
    v8bf ahi = *(const v8bf*)(pa + 16);  // K = k0+kh+16 .. +23
    const __bf16* pb = Wt + (size_t)colB * K + k0 + kb;
    v8bf blo = *(const v8bf*)pb;         // K = k0+kb .. +7
    v8bf bhi = *(const v8bf*)(pb + 16);  // K = k0+kb+16 .. +23
    v16bf a, b;
#pragma unroll
    for (int i = 0; i < 8; ++i) {
      a[i] = alo[i];
      a[i + 8] = ahi[i];
      b[i] = blo[i];
      b[i + 8] = bhi[i];
    }
    acc = __builtin_amdgcn_wmma_f32_16x16x32_bf16(false, a, false, b,
                                                  (short)0, acc, false, false);
  }
  int mb = m0 + ((lane >> 4) << 3);  // C/D: VGPR r -> M = r (+8 for hi lanes)
  int n = n0 + (lane & 15);
#pragma unroll
  for (int r = 0; r < 8; ++r) H[(size_t)(mb + r) * Co + n] = acc[r];
}

// --------------------------- edge aggregation ------------------------------
// agg[col[e], c] += norm[e] * H[row[e], c]   (self-loops at e in [E, E+N))
__global__ void edge_scatter_k(const long long* __restrict__ ei,
                               const float* __restrict__ nrm,
                               const float* __restrict__ H,
                               float* __restrict__ agg, int E, int N, int C) {
  long long t = (long long)blockIdx.x * blockDim.x + threadIdx.x;
  long long total = (long long)(E + N) * C;
  if (t >= total) return;
  int c = (int)(t % C);
  long long e = t / C;
  int r, cl;
  if (e < E) {
    r = (int)ei[e];
    cl = (int)ei[(long long)E + e];
  } else {
    r = cl = (int)(e - E);
  }
  atomicAdd(&agg[(size_t)cl * C + c], nrm[e] * H[(size_t)r * C + c]);
}

// --------------------------------- LSTM ------------------------------------
// One block per graph, 128 threads (one per hidden unit). Gate order i,f,g,o.
__global__ __launch_bounds__(128) void lstm_k(const float* __restrict__ inp, int D,
                                              float* __restrict__ h, float* __restrict__ c,
                                              const float* __restrict__ Wih,
                                              const float* __restrict__ Whh,
                                              const float* __restrict__ bih,
                                              const float* __restrict__ bhh) {
  __shared__ float sI[256];
  __shared__ float sH[128];
  int g = blockIdx.x, j = threadIdx.x;
  for (int i = j; i < D; i += 128) sI[i] = inp[(size_t)g * D + i];
  sH[j] = h[(size_t)g * 128 + j];
  __syncthreads();
  float acc[4];
#pragma unroll
  for (int q = 0; q < 4; ++q) acc[q] = bih[q * 128 + j] + bhh[q * 128 + j];
#pragma unroll
  for (int q = 0; q < 4; ++q) {
    const float* wi = Wih + (size_t)(q * 128 + j) * D;
    const float* wh = Whh + (size_t)(q * 128 + j) * 128;
    float s = 0.f;
    for (int d = 0; d < D; ++d) s += wi[d] * sI[d];
    for (int d = 0; d < 128; ++d) s += wh[d] * sH[d];
    acc[q] += s;
  }
  float cNew = sigmoidf_(acc[1]) * c[(size_t)g * 128 + j] + sigmoidf_(acc[0]) * tanhf(acc[2]);
  float hNew = sigmoidf_(acc[3]) * tanhf(cNew);
  c[(size_t)g * 128 + j] = cNew;
  h[(size_t)g * 128 + j] = hNew;
}

// ------------------------- set2set attention -------------------------------
// One wave per node: e[n] = dot(x[n], q[batch[n]]); atomic segment max.
__global__ void att_logits_k(const float* __restrict__ x, const float* __restrict__ q,
                             const long long* __restrict__ batch, float* __restrict__ e,
                             unsigned* __restrict__ emax, int N, int C) {
  int wid = (blockIdx.x * blockDim.x + threadIdx.x) >> 5;
  int lane = threadIdx.x & 31;
  if (wid >= N) return;
  int g = (int)batch[wid];
  const float* xr = x + (size_t)wid * C;
  const float* qr = q + (size_t)g * C;
  float s = 0.f;
  for (int c = lane; c < C; c += 32) s += xr[c] * qr[c];
  for (int off = 16; off; off >>= 1) s += __shfl_down(s, off);
  if (lane == 0) {
    e[wid] = s;
    atomicMax(&emax[g], fOrd(s));
  }
}
__global__ void att_exp_k(const float* __restrict__ e, const unsigned* __restrict__ emax,
                          const long long* __restrict__ batch, float* __restrict__ a,
                          float* __restrict__ denom, int N) {
  int i = blockIdx.x * blockDim.x + threadIdx.x;
  if (i >= N) return;
  int g = (int)batch[i];
  float v = expf(e[i] - fUnord(emax[g]));
  a[i] = v;
  atomicAdd(&denom[g], v);
}
__global__ void att_r_k(const float* __restrict__ a, const float* __restrict__ denom,
                        const long long* __restrict__ batch, const float* __restrict__ x,
                        float* __restrict__ r, long long total, int C) {
  long long t = (long long)blockIdx.x * blockDim.x + threadIdx.x;
  if (t >= total) return;
  int c = (int)(t % C);
  long long n = t / C;
  int g = (int)batch[n];
  atomicAdd(&r[(size_t)g * C + c], (a[n] / denom[g]) * x[(size_t)n * C + c]);
}
__global__ void qstar_k(const float* __restrict__ q, const float* __restrict__ r,
                        float* __restrict__ qs) {
  int i = blockIdx.x * blockDim.x + threadIdx.x;
  if (i >= 64 * 256) return;
  int g = i >> 8, c = i & 255;
  qs[i] = (c < 128) ? q[g * 128 + c] : r[g * 128 + (c - 128)];
}

// ------------------------------ DNN head -----------------------------------
__global__ void small_gemm_k(const float* __restrict__ A, const float* __restrict__ W,
                             const float* __restrict__ bias, float* __restrict__ out,
                             int B, int D, int O) {
  int i = blockIdx.x * blockDim.x + threadIdx.x;
  if (i >= B * O) return;
  int g = i / O, o = i % O;
  float s = bias ? bias[o] : 0.f;
  const float* a = A + (size_t)g * D;
  const float* w = W + (size_t)o * D;
  for (int d = 0; d < D; ++d) s += a[d] * w[d];
  out[i] = s;
}
__global__ void bn_elu_small_k(float* __restrict__ X, const float* __restrict__ gamma,
                               const float* __restrict__ beta, int B, int O) {
  int o = threadIdx.x;
  if (o >= O) return;
  float s = 0.f, q = 0.f;
  for (int g = 0; g < B; ++g) {
    float v = X[g * O + o];
    s += v;
    q += v * v;
  }
  float mu = s / B;
  float var = q / B - mu * mu;
  float sc = gamma[o] * rsqrtf(var + 1e-5f);
  float sh = beta[o] - mu * sc;
  for (int g = 0; g < B; ++g) {
    float v = X[g * O + o] * sc + sh;
    X[g * O + o] = v > 0.f ? v : expm1f(v);
  }
}

// ----------------------------- orchestration -------------------------------
struct Ws {
  float* nrm;
  float* deg;
  float* dinv;
  float* sums;   // 256 floats: sum[0..127], sq[128..255]
  float* scale;
  float* shift;
  __bf16* Wt;
  __bf16* Xbf;
  float* H;
  float* agg;
  float* hfeat;
  float* hS;
  float* cS;
  float* qs;
  float* evec;
  float* avec;
  unsigned* emax;
  float* denom;
  float* rbuf;
  float* o1;
};

static void run_branch(void* const* d_in, int base, int nLstm,
                       const float* x, const long long* ei, const long long* batch,
                       int N, int E, const Ws& W, float* outPtr, hipStream_t stream) {
  // JAX pytree (sorted-key) leaf order within a branch:
  // bn0(g,b) bn1 bn2 bn3 | dnn1(W,b) dnn2(W,b) dnn_bn(g,b) |
  // gcn1(W,b) gcn2 gcn3 | lstm[l]: W_hh, W_ih, b_hh, b_ih
  auto P = [&](int i) { return (const float*)d_in[base + i]; };
  const float* bnG[4] = {P(0), P(2), P(4), P(6)};
  const float* bnB[4] = {P(1), P(3), P(5), P(7)};
  const float* dnn1W = P(8);
  const float* dnn2W = P(10);
  const float* dnn2b = P(11);
  const float* dbnG = P(12);
  const float* dbnB = P(13);
  const float* gW[3] = {P(14), P(16), P(18)};

  // ---- bn0 on x (C=128) -> Xbf ----
  fill_f_k<<<1, 256, 0, stream>>>(W.sums, 0.f, 256);
  bn_stats_k<<<CDIV(N, 512), 128, 0, stream>>>(x, N, 128, W.sums, W.sums + 128, 512);
  bn_finalize_k<<<1, 128, 0, stream>>>(W.sums, W.sums + 128, bnG[0], bnB[0],
                                       W.scale, W.shift, N, 128);
  bn_apply_k<<<CDIV((long long)N * 128, 256), 256, 0, stream>>>(
      x, W.scale, W.shift, W.Xbf, nullptr, (long long)N * 128, 128, 0);

  // ---- 3 GCN layers ----
  const int Ks[3] = {128, 32, 64};
  const int Cos[3] = {32, 64, 128};
  for (int i = 0; i < 3; ++i) {
    int K = Ks[i], Co = Cos[i];
    conv_wt_k<<<CDIV(K * Co, 256), 256, 0, stream>>>(gW[i], W.Wt, K, Co);
    long long tiles = (long long)(N >> 4) * (Co >> 4);
    wmma_mm_k<<<(unsigned)CDIV(tiles, 8), 256, 0, stream>>>(W.Xbf, W.Wt, W.H, N, K, Co);
    fill_f_k<<<(unsigned)CDIV((long long)N * Co, 256), 256, 0, stream>>>(
        W.agg, 0.f, (long long)N * Co);
    edge_scatter_k<<<(unsigned)CDIV((long long)(E + N) * Co, 256), 256, 0, stream>>>(
        ei, W.nrm, W.H, W.agg, E, N, Co);
    fill_f_k<<<1, 256, 0, stream>>>(W.sums, 0.f, 256);
    bn_stats_k<<<CDIV(N, 512), Co, 0, stream>>>(W.agg, N, Co, W.sums, W.sums + 128, 512);
    bn_finalize_k<<<1, Co, 0, stream>>>(W.sums, W.sums + 128, bnG[i + 1], bnB[i + 1],
                                        W.scale, W.shift, N, Co);
    bn_apply_k<<<(unsigned)CDIV((long long)N * Co, 256), 256, 0, stream>>>(
        W.agg, W.scale, W.shift, (i < 2) ? W.Xbf : nullptr,
        (i == 2) ? W.hfeat : nullptr, (long long)N * Co, Co, 1);
  }

  // ---- Set2Set (4 steps) ----
  fill_f_k<<<CDIV(nLstm * 64 * 128, 256), 256, 0, stream>>>(W.hS, 0.f, nLstm * 64 * 128);
  fill_f_k<<<CDIV(nLstm * 64 * 128, 256), 256, 0, stream>>>(W.cS, 0.f, nLstm * 64 * 128);
  fill_f_k<<<CDIV(64 * 256, 256), 256, 0, stream>>>(W.qs, 0.f, 64 * 256);
  for (int step = 0; step < 4; ++step) {
    const float* inp = W.qs;
    int D = 256;
    for (int l = 0; l < nLstm; ++l) {
      const float* Whh = P(20 + 4 * l);
      const float* Wih = P(21 + 4 * l);
      const float* bhh = P(22 + 4 * l);
      const float* bih = P(23 + 4 * l);
      lstm_k<<<64, 128, 0, stream>>>(inp, D, W.hS + l * 8192, W.cS + l * 8192,
                                     Wih, Whh, bih, bhh);
      inp = W.hS + l * 8192;
      D = 128;
    }
    const float* q = W.hS + (nLstm - 1) * 8192;
    fill_u_k<<<1, 64, 0, stream>>>(W.emax, 0u, 64);
    fill_f_k<<<1, 64, 0, stream>>>(W.denom, 0.f, 64);
    fill_f_k<<<CDIV(64 * 128, 256), 256, 0, stream>>>(W.rbuf, 0.f, 64 * 128);
    att_logits_k<<<(unsigned)CDIV((long long)N * 32, 256), 256, 0, stream>>>(
        W.hfeat, q, batch, W.evec, W.emax, N, 128);
    att_exp_k<<<CDIV(N, 256), 256, 0, stream>>>(W.evec, W.emax, batch, W.avec, W.denom, N);
    att_r_k<<<(unsigned)CDIV((long long)N * 128, 256), 256, 0, stream>>>(
        W.avec, W.denom, batch, W.hfeat, W.rbuf, (long long)N * 128, 128);
    qstar_k<<<CDIV(64 * 256, 256), 256, 0, stream>>>(q, W.rbuf, W.qs);
  }

  // ---- DNN head (dnn1 bias absorbed by train-mode BN) ----
  small_gemm_k<<<CDIV(64 * 64, 256), 256, 0, stream>>>(W.qs, dnn1W, nullptr, W.o1, 64, 256, 64);
  bn_elu_small_k<<<1, 64, 0, stream>>>(W.o1, dbnG, dbnB, 64, 64);
  small_gemm_k<<<CDIV(64 * 32, 256), 256, 0, stream>>>(W.o1, dnn2W, dnn2b, outPtr, 64, 64, 32);
}

extern "C" void kernel_launch(void* const* d_in, const int* in_sizes, int n_in,
                              void* d_out, int out_size, void* d_ws, size_t ws_size,
                              hipStream_t stream) {
  const float* x = (const float*)d_in[0];
  const long long* ei = (const long long*)d_in[1];     // int64 (2, E) flat
  const float* ew = (const float*)d_in[2];
  const long long* batch = (const long long*)d_in[3];  // int64 (N,), sorted

  const int N = in_sizes[0] / 128;
  const int E = in_sizes[1] / 2;

  // ---- carve workspace (aligned to 256B) ----
  char* cur = (char*)d_ws;
  auto take = [&](size_t bytes) {
    char* p = cur;
    cur += (bytes + 255) & ~(size_t)255;
    return (void*)p;
  };
  Ws W;
  W.nrm = (float*)take(sizeof(float) * (size_t)(E + N));
  W.deg = (float*)take(sizeof(float) * N);
  W.dinv = (float*)take(sizeof(float) * N);
  W.sums = (float*)take(sizeof(float) * 256);
  W.scale = (float*)take(sizeof(float) * 128);
  W.shift = (float*)take(sizeof(float) * 128);
  W.Wt = (__bf16*)take(sizeof(__bf16) * 128 * 128);
  W.Xbf = (__bf16*)take(sizeof(__bf16) * (size_t)N * 128);
  W.H = (float*)take(sizeof(float) * (size_t)N * 128);
  W.agg = (float*)take(sizeof(float) * (size_t)N * 128);
  W.hfeat = (float*)take(sizeof(float) * (size_t)N * 128);
  W.hS = (float*)take(sizeof(float) * 2 * 64 * 128);
  W.cS = (float*)take(sizeof(float) * 2 * 64 * 128);
  W.qs = (float*)take(sizeof(float) * 64 * 256);
  W.evec = (float*)take(sizeof(float) * N);
  W.avec = (float*)take(sizeof(float) * N);
  W.emax = (unsigned*)take(sizeof(unsigned) * 64);
  W.denom = (float*)take(sizeof(float) * 64);
  W.rbuf = (float*)take(sizeof(float) * 64 * 128);
  W.o1 = (float*)take(sizeof(float) * 64 * 64);
  (void)ws_size;

  // ---- gcn_norm (shared by both branches) ----
  fill_f_k<<<CDIV(N, 256), 256, 0, stream>>>(W.deg, 1.0f, N);  // self-loop weight 1
  deg_edges_k<<<CDIV(E, 256), 256, 0, stream>>>(ei, ew, W.deg, E);
  dinv_k<<<CDIV(N, 256), 256, 0, stream>>>(W.deg, W.dinv, N);
  norm_k<<<CDIV(E + N, 256), 256, 0, stream>>>(ei, ew, W.dinv, W.nrm, E, N);

  // ---- branches ----
  float* out = (float*)d_out;
  run_branch(d_in, /*base=*/4, /*nLstm=*/1, x, ei, batch, N, E, W, out, stream);
  run_branch(d_in, /*base=*/28, /*nLstm=*/2, x, ei, batch, N, E, W, out + 64 * 32, stream);
}
